// AutoDecoderLayer_49366354100291
// MI455X (gfx1250) — matrix operand order repaired and checked by reading, not compiled, not run on hardware
//
#include <hip/hip_runtime.h>
#include <hip/hip_bf16.h>

// ---------------------------------------------------------------------------
// AutoDecoderLayer on gfx1250 (MI455X): bf16 WMMA GEMMs + f32 softmax/LN glue.
// Round 3: TDM (tensor_load_to_lds) staging of the A tile (wave-0 issued DMA,
// s_wait_tensorcnt handshake), overlapped with manual swizzled B staging.
// ---------------------------------------------------------------------------

#define USE_TDM 1

typedef __attribute__((ext_vector_type(16))) __bf16          v16bf;
typedef __attribute__((ext_vector_type(8)))  float           v8f;
typedef __attribute__((ext_vector_type(4)))  unsigned int    u32x4;
typedef __attribute__((ext_vector_type(4)))  unsigned int    tdm_g0_t;  // D# group0
typedef __attribute__((ext_vector_type(8)))  int             tdm_g1_t;  // D# group1
typedef __attribute__((ext_vector_type(4)))  int             tdm_g2_t;  // D# group2/3

struct frag32 { u32x4 lo, hi; };   // 32 bytes = one 16-element bf16 fragment

#define SEQ   4096
#define EMB   1024
#define HEADS 8
#define HE    (EMB * HEADS)

__device__ __forceinline__ unsigned short f2bf(float f) {
    unsigned int u = __float_as_uint(f);
    unsigned int r = u + 0x7FFFu + ((u >> 16) & 1u);   // round-to-nearest-even
    return (unsigned short)(r >> 16);
}

__device__ __forceinline__ unsigned long long pack4bf(float a, float b, float c, float d) {
    return (unsigned long long)f2bf(a)
         | ((unsigned long long)f2bf(b) << 16)
         | ((unsigned long long)f2bf(c) << 32)
         | ((unsigned long long)f2bf(d) << 48);
}

// ---------------------------------------------------------------------------
// f32 -> bf16 conversion, float4 in / 8B packed out (n % 4 == 0 always here)
// ---------------------------------------------------------------------------
__global__ __launch_bounds__(256) void f32_to_bf16_kernel(
    const float* __restrict__ in, unsigned short* __restrict__ out, long long n)
{
    long long n4 = n >> 2;
    long long i = (long long)blockIdx.x * 256 + threadIdx.x;
    long long stride = (long long)gridDim.x * 256;
    const float4* in4 = (const float4*)in;
    unsigned long long* out4 = (unsigned long long*)out;
    for (; i < n4; i += stride) {
        float4 v = in4[i];
        out4[i] = pack4bf(v.x, v.y, v.z, v.w);
    }
}

// ---------------------------------------------------------------------------
// WMMA GEMM:  C[M,N] = scale * opA(A) x opB(B) + bias
//   TA=0: logical A[i,k] = A[i*lda + k]      TA=1: A[i,k] = A[k*lda + i]
//   TB=0: logical B[k,j] = B[k*ldb + j]      TB=1: B[k,j] = B[j*ldb + k]
// Block = 256 thr = 8 wave32 (4x2 waves); wave tile 32x64 = 2x4 wmma tiles.
// A tile: TDM DMA (TA=0) into row-major LDS, rows padded to 144B via the D#
//         pad feature (pad_interval=4 -> every 32 DWORDs; pad_amount=3 -> 16B).
// B tile: pre-swizzled into per-lane fragment order -> contiguous 32B reads.
// ---------------------------------------------------------------------------
#define BM 128
#define BN 128
#define BK 64
#define APAD 8    // ushorts of row padding: row stride 72*2 = 144 bytes

template<bool TA, bool TB>
__global__ __launch_bounds__(256) void gemm_wmma_bf16(
    const unsigned short* __restrict__ A,
    const unsigned short* __restrict__ B,
    float* __restrict__ Cf,
    unsigned short* __restrict__ Cb,
    const float* __restrict__ bias,
    int M, int N, int K,
    int lda, int ldb, int ldc,
    long long sA, long long sB, long long sC, long long sBias,
    float scale)
{
    __shared__ unsigned short As[BM][BK + APAD];
    __shared__ unsigned short Bs[2][BN / 16][32][16];   // [kHalf][nTile][lane][elem]

    const int bz = blockIdx.z;
    A += (long long)bz * sA;
    B += (long long)bz * sB;
    if (Cf)   Cf   += (long long)bz * sC;
    if (Cb)   Cb   += (long long)bz * sC;
    if (bias) bias += (long long)bz * sBias;

    const int i0   = blockIdx.y * BM;
    const int j0   = blockIdx.x * BN;
    const int tid  = threadIdx.x;
    const int lane = tid & 31;
    const int wid  = tid >> 5;
    const int mBase = (wid & 3) * 32;   // 4 waves along M
    const int nBase = (wid >> 2) * 64;  // 2 waves along N
    const int lm  = lane & 15;
    const int lhi = lane >> 4;

    v8f zero = {0.f, 0.f, 0.f, 0.f, 0.f, 0.f, 0.f, 0.f};
    v8f acc[2][4];
#pragma unroll
    for (int tm = 0; tm < 2; ++tm)
#pragma unroll
        for (int tn = 0; tn < 4; ++tn) acc[tm][tn] = zero;

#if USE_TDM
    // Invariant parts of the Tensor DMA descriptor (ISA 8.3/8.4), TA=0 path.
    unsigned int ldsAddr = 0;
    tdm_g1_t g1 = {0, 0, 0, 0, 0, 0, 0, 0};
    if (!TA) {
        ldsAddr = (unsigned int)(unsigned long long)(uintptr_t)&As[0][0];
        // group1 word0: data_size=2B (bits17:16=1), pad_enable (bit20),
        //               pad_interval=4 (bits24:22, pad after 32 DWORDs=128B),
        //               pad_amount=3 (bits31:25, 4 DWORDs=16B)
        g1[0] = (int)((1u << 16) | (1u << 20) | (4u << 22) | (3u << 25));
        g1[1] = (int)((unsigned)BK << 16);   // tensor_dim0[15:0]  (bits 79:64 lo half)
        g1[2] = (int)((unsigned)BM << 16);   // tensor_dim1[15:0]  (bits 111:96 lo half)
        g1[3] = (int)((unsigned)BK << 16);   // tile_dim0          (bits 127:112)
        g1[4] = (int)((unsigned)BM);         // tile_dim1 (tile_dim2=0)
        g1[5] = (int)((unsigned)lda);        // tensor_dim0_stride[31:0]
        g1[6] = 0;                           // stride[47:32]=0, dim1_stride lo=0
        g1[7] = 0;
    }
    const tdm_g2_t gz = {0, 0, 0, 0};        // 2-D tensor: groups 2/3 unused
#endif

    for (int k0 = 0; k0 < K; k0 += BK) {
        // ================= stage A tile (BM x BK) =================
        if (!TA) {
#if USE_TDM
            if (wid == 0) {   // one DMA per block; TDM ignores EXEC, is per-wave
                unsigned long long ga =
                    (unsigned long long)(uintptr_t)(A + (long long)i0 * lda + k0);
                tdm_g0_t g0;
                g0[0] = 1u;                                   // count=1, user mode
                g0[1] = ldsAddr;                              // LDS dest (bytes)
                g0[2] = (unsigned int)(ga & 0xFFFFFFFFull);   // global_addr[31:0]
                g0[3] = (unsigned int)((ga >> 32) & 0x1FFFFFFull)
                      | (2u << 30);                           // addr[56:32] | type=2
#if __clang_major__ >= 23
                tdm_g1_t gz8 = {0, 0, 0, 0, 0, 0, 0, 0};
                __builtin_amdgcn_tensor_load_to_lds(g0, g1, gz, gz, gz8, 0);
#else
                __builtin_amdgcn_tensor_load_to_lds(g0, g1, gz, gz, 0);
#endif
            }
#else
            const int row  = tid >> 1;
            const int half = tid & 1;            // 32-element (64B) half of the row
            const u32x4* src = (const u32x4*)(A + (long long)(i0 + row) * lda + k0 + half * 32);
            u32x4 v0 = src[0], v1 = src[1], v2 = src[2], v3 = src[3];
            u32x4* dst = (u32x4*)&As[row][half * 32];
            dst[0] = v0; dst[1] = v1; dst[2] = v2; dst[3] = v3;
            if (k0 + BK < K) __builtin_prefetch((const char*)(src) + BK * 2, 0, 1);
#endif
        } else {
            const int k  = tid >> 2;             // 0..63
            const int mb = (tid & 3) * 32;
            const unsigned short* src = A + (long long)(k0 + k) * lda + (i0 + mb);
#pragma unroll
            for (int e = 0; e < 32; ++e) As[mb + e][k] = src[e];
            if (k0 + BK < K) __builtin_prefetch((const char*)(src + (long long)BK * lda), 0, 1);
        }
        // ================= stage B tile (BK x BN) =================
        if (!TB) {
            const int k  = tid >> 2;             // 0..63
            const int jb = (tid & 3) * 32;
            const unsigned short* src = B + (long long)(k0 + k) * ldb + j0 + jb;
            const int kk2 = k >> 5;
            const int laneAdd = ((k & 31) >> 4) << 4;
            const int elem = k & 15;
#pragma unroll
            for (int e = 0; e < 32; ++e)
                Bs[kk2][(jb >> 4) + (e >> 4)][(e & 15) + laneAdd][elem] = src[e];
            if (k0 + BK < K) __builtin_prefetch((const char*)(src + (long long)BK * ldb), 0, 1);
        } else {
            const int j    = tid >> 1;           // 0..127
            const int half = tid & 1;            // k half: 0 -> k 0..31, 1 -> k 32..63
            const u32x4* src = (const u32x4*)(B + (long long)(j0 + j) * ldb + k0 + half * 32);
            u32x4 v0 = src[0], v1 = src[1], v2 = src[2], v3 = src[3];
            u32x4* d0 = (u32x4*)&Bs[half][j >> 4][(j & 15)][0];
            u32x4* d1 = (u32x4*)&Bs[half][j >> 4][(j & 15) + 16][0];
            d0[0] = v0; d0[1] = v1;              // k%32 in [0,16)  -> lanes 0-15
            d1[0] = v2; d1[1] = v3;              // k%32 in [16,32) -> lanes 16-31
            if (k0 + BK < K) __builtin_prefetch((const char*)(src) + BK * 2, 0, 1);
        }
#if USE_TDM
        if (!TA && wid == 0) __builtin_amdgcn_s_wait_tensorcnt(0);
#endif
        __syncthreads();

        // ============ 2 WMMA K-steps per staging round ============
#pragma unroll
        for (int kk = 0; kk < BK; kk += 32) {
            // A fragment: one contiguous 32B run per lane:
            //   elems 0-15 -> K = kk + 16*lhi + 0..15
            frag32 fa[2];
#pragma unroll
            for (int tm = 0; tm < 2; ++tm) {
                const u32x4* ap = (const u32x4*)((const char*)&As[mBase + tm * 16 + lm][kk]
                                                 + lhi * 32);
                fa[tm].lo = ap[0];
                fa[tm].hi = ap[1];               // +16 bytes
            }
            // B fragment: one contiguous 32B chunk per lane (pre-swizzled)
            frag32 fb[4];
#pragma unroll
            for (int tn = 0; tn < 4; ++tn) {
                const u32x4* bp = (const u32x4*)&Bs[kk >> 5][(nBase >> 4) + tn][lane][0];
                fb[tn].lo = bp[0];
                fb[tn].hi = bp[1];
            }
#pragma unroll
            for (int tm = 0; tm < 2; ++tm)
#pragma unroll
                for (int tn = 0; tn < 4; ++tn)
                    acc[tm][tn] = __builtin_amdgcn_wmma_f32_16x16x32_bf16(
                        false, __builtin_bit_cast(v16bf, fa[tm]),
                        false, __builtin_bit_cast(v16bf, fb[tn]),
                        (short)0, acc[tm][tn], false, false);
        }
        __syncthreads();
    }

    // ---- epilogue: C layout lane -> N = lm, VGPR r -> M = r + 8*lhi ----
#pragma unroll
    for (int tm = 0; tm < 2; ++tm) {
#pragma unroll
        for (int tn = 0; tn < 4; ++tn) {
            int gn = j0 + nBase + tn * 16 + lm;
            float bv = bias ? bias[gn] : 0.0f;
#pragma unroll
            for (int r = 0; r < 8; ++r) {
                int gm = i0 + mBase + tm * 16 + lhi * 8 + r;
                float v = acc[tm][tn][r] * scale + bv;
                long long idx = (long long)gm * ldc + gn;
                if (Cf) Cf[idx] = v;
                if (Cb) Cb[idx] = f2bf(v);
            }
        }
    }
}

// ---------------------------------------------------------------------------
// Row softmax: f32 scores -> bf16 probabilities. One block per 1024-col row.
// Each of the 256 threads owns exactly one float4.
// ---------------------------------------------------------------------------
__global__ __launch_bounds__(256) void softmax_rows_kernel(
    const float* __restrict__ S, unsigned short* __restrict__ P)
{
    const long long row = blockIdx.x;
    const float4* src = (const float4*)(S + row * EMB);
    unsigned long long* dst = (unsigned long long*)(P + row * EMB);
    __shared__ float red[256];
    const int tid = threadIdx.x;

    float4 v = src[tid];
    float mx = fmaxf(fmaxf(v.x, v.y), fmaxf(v.z, v.w));
    red[tid] = mx; __syncthreads();
#pragma unroll
    for (int s = 128; s > 0; s >>= 1) {
        if (tid < s) red[tid] = fmaxf(red[tid], red[tid + s]);
        __syncthreads();
    }
    mx = red[0]; __syncthreads();

    float e0 = __expf(v.x - mx), e1 = __expf(v.y - mx);
    float e2 = __expf(v.z - mx), e3 = __expf(v.w - mx);
    red[tid] = e0 + e1 + e2 + e3; __syncthreads();
#pragma unroll
    for (int s = 128; s > 0; s >>= 1) {
        if (tid < s) red[tid] += red[tid + s];
        __syncthreads();
    }
    float inv = 1.0f / red[0];
    dst[tid] = pack4bf(e0 * inv, e1 * inv, e2 * inv, e3 * inv);
}

// ---------------------------------------------------------------------------
// LayerNorm + residual: out = LN(x)*g + b + res. One block per 1024-col row.
// ---------------------------------------------------------------------------
__global__ __launch_bounds__(256) void layernorm_res_kernel(
    const float* __restrict__ X, const float* __restrict__ g,
    const float* __restrict__ b, const float* __restrict__ res,
    float* __restrict__ outF, unsigned short* __restrict__ outB)
{
    const long long row = blockIdx.x;
    const float4* x4 = (const float4*)(X + row * EMB);
    const float4* r4 = (const float4*)(res + row * EMB);
    __shared__ float red[256];
    const int tid = threadIdx.x;

    float4 v = x4[tid];
    float sum = v.x + v.y + v.z + v.w;
    float sq  = v.x * v.x + v.y * v.y + v.z * v.z + v.w * v.w;
    red[tid] = sum; __syncthreads();
#pragma unroll
    for (int s = 128; s > 0; s >>= 1) { if (tid < s) red[tid] += red[tid + s]; __syncthreads(); }
    float mean = red[0] * (1.0f / EMB); __syncthreads();
    red[tid] = sq; __syncthreads();
#pragma unroll
    for (int s = 128; s > 0; s >>= 1) { if (tid < s) red[tid] += red[tid + s]; __syncthreads(); }
    float var  = red[0] * (1.0f / EMB) - mean * mean;
    float rstd = rsqrtf(var + 1e-5f);

    const float4 gv = ((const float4*)g)[tid];
    const float4 bv = ((const float4*)b)[tid];
    const float4 rv = r4[tid];
    float y0 = (v.x - mean) * rstd * gv.x + bv.x + rv.x;
    float y1 = (v.y - mean) * rstd * gv.y + bv.y + rv.y;
    float y2 = (v.z - mean) * rstd * gv.z + bv.z + rv.z;
    float y3 = (v.w - mean) * rstd * gv.w + bv.w + rv.w;
    if (outF) {
        float4 o; o.x = y0; o.y = y1; o.z = y2; o.w = y3;
        ((float4*)(outF + row * EMB))[tid] = o;
    }
    if (outB)
        ((unsigned long long*)(outB + row * EMB))[tid] = pack4bf(y0, y1, y2, y3);
}

// ---------------------------------------------------------------------------
// Host launch
// ---------------------------------------------------------------------------
extern "C" void kernel_launch(void* const* d_in, const int* in_sizes, int n_in,
                              void* d_out, int out_size, void* d_ws, size_t ws_size,
                              hipStream_t stream) {
    (void)in_sizes; (void)n_in; (void)out_size; (void)ws_size;

    const float* history = (const float*)d_in[0];   // [S,E]
    const float* embd    = (const float*)d_in[1];   // [S,E]
    const float* Wq_w    = (const float*)d_in[2];   // [H,E,E]
    const float* Wq_b    = (const float*)d_in[3];   // [H,E]
    const float* Wk_w    = (const float*)d_in[4];
    const float* Wk_b    = (const float*)d_in[5];
    const float* Wv_w    = (const float*)d_in[6];
    const float* Wv_b    = (const float*)d_in[7];
    const float* Wz_w    = (const float*)d_in[8];   // [E, H*E]
    const float* Wz_b    = (const float*)d_in[9];   // [E]
    const float* ln1_g   = (const float*)d_in[10];
    const float* ln1_b   = (const float*)d_in[11];
    const float* Wf_w    = (const float*)d_in[12];  // [E,E]
    const float* Wf_b    = (const float*)d_in[13];
    const float* ln2_g   = (const float*)d_in[14];
    const float* ln2_b   = (const float*)d_in[15];

    // workspace carve-up (256B aligned)
    char* w = (char*)d_ws;
    auto take = [&](size_t bytes) -> void* {
        void* p = (void*)w;
        w += (bytes + 255) & ~(size_t)255;
        return p;
    };
    const long long SE  = (long long)SEQ * EMB;       // 4M
    const long long EE  = (long long)EMB * EMB;       // 1M
    const long long HEE = (long long)HEADS * EE;      // 8M
    const long long SHE = (long long)SEQ * HE;        // 32M

    unsigned short* Xb   = (unsigned short*)take(SE  * 2);
    unsigned short* Hb   = (unsigned short*)take(SE  * 2);
    unsigned short* Wqb  = (unsigned short*)take(HEE * 2);
    unsigned short* Wkb  = (unsigned short*)take(HEE * 2);
    unsigned short* Wvb  = (unsigned short*)take(HEE * 2);
    unsigned short* Wzb  = (unsigned short*)take((long long)EMB * HE * 2);
    unsigned short* Wfb  = (unsigned short*)take(EE  * 2);
    unsigned short* Qb   = (unsigned short*)take((long long)HEADS * SE * 2);
    unsigned short* Kb   = (unsigned short*)take((long long)HEADS * SE * 2);
    unsigned short* Vb   = (unsigned short*)take((long long)HEADS * SE * 2);
    float*          Sf   = (float*)         take(HEE * 4);
    unsigned short* Ab   = (unsigned short*)take(HEE * 2);
    unsigned short* Zb   = (unsigned short*)take(SHE * 2);
    float*          Of   = (float*)         take(SE  * 4);
    float*          LN1f = (float*)         take(SE  * 4);
    unsigned short* LN1b = (unsigned short*)take(SE  * 2);
    float*          FNf  = (float*)         take(SE  * 4);

    // ---- f32 -> bf16 conversions ----
    f32_to_bf16_kernel<<<2048, 256, 0, stream>>>(embd,    Xb,  SE);
    f32_to_bf16_kernel<<<2048, 256, 0, stream>>>(history, Hb,  SE);
    f32_to_bf16_kernel<<<2048, 256, 0, stream>>>(Wq_w,    Wqb, HEE);
    f32_to_bf16_kernel<<<2048, 256, 0, stream>>>(Wk_w,    Wkb, HEE);
    f32_to_bf16_kernel<<<2048, 256, 0, stream>>>(Wv_w,    Wvb, HEE);
    f32_to_bf16_kernel<<<2048, 256, 0, stream>>>(Wz_w,    Wzb, (long long)EMB * HE);
    f32_to_bf16_kernel<<<2048, 256, 0, stream>>>(Wf_w,    Wfb, EE);

    dim3 blk(256);
    // ---- Q/K/V per head: [S,E] x [E,E]^T + b  (batched over heads in z) ----
    {
        dim3 grid(EMB / BN, SEQ / BM, HEADS);
        gemm_wmma_bf16<false, true><<<grid, blk, 0, stream>>>(
            Xb, Wqb, nullptr, Qb, Wq_b,
            SEQ, EMB, EMB, EMB, EMB, EMB,
            0LL, EE, SE, (long long)EMB, 1.0f);
        gemm_wmma_bf16<false, true><<<grid, blk, 0, stream>>>(
            Xb, Wkb, nullptr, Kb, Wk_b,
            SEQ, EMB, EMB, EMB, EMB, EMB,
            0LL, EE, SE, (long long)EMB, 1.0f);
        gemm_wmma_bf16<false, true><<<grid, blk, 0, stream>>>(
            Hb, Wvb, nullptr, Vb, Wv_b,
            SEQ, EMB, EMB, EMB, EMB, EMB,
            0LL, EE, SE, (long long)EMB, 1.0f);
    }
    // ---- scores = K^T Q / sqrt(E): [E,S]x[S,E] per head ----
    {
        dim3 grid(EMB / BN, EMB / BM, HEADS);
        gemm_wmma_bf16<true, false><<<grid, blk, 0, stream>>>(
            Kb, Qb, Sf, nullptr, nullptr,
            EMB, EMB, SEQ, EMB, EMB, EMB,
            SE, SE, EE, 0LL, 0.03125f /* 1/sqrt(1024) */);
    }
    // ---- softmax over last axis of [H,E,E] ----
    softmax_rows_kernel<<<HEADS * EMB, blk, 0, stream>>>(Sf, Ab);
    // ---- Z_h = V A : [S,E]x[E,E]; write into concatenated Z[s, h*E+f] ----
    {
        dim3 grid(EMB / BN, SEQ / BM, HEADS);
        gemm_wmma_bf16<false, false><<<grid, blk, 0, stream>>>(
            Vb, Ab, nullptr, Zb, nullptr,
            SEQ, EMB, EMB, EMB, EMB, HE /* ldc spans all heads */,
            SE, EE, (long long)EMB /* column offset per head */, 0LL, 1.0f);
    }
    // ---- O = Z Wz^T + bz : [S,8192] x [E,8192]^T ----
    {
        dim3 grid(EMB / BN, SEQ / BM, 1);
        gemm_wmma_bf16<false, true><<<grid, blk, 0, stream>>>(
            Zb, Wzb, Of, nullptr, Wz_b,
            SEQ, EMB, HE, HE, HE, EMB,
            0LL, 0LL, 0LL, 0LL, 1.0f);
    }
    // ---- LN1 = layernorm(O)*g+b + embd ----
    layernorm_res_kernel<<<SEQ, blk, 0, stream>>>(Of, ln1_g, ln1_b, embd, LN1f, LN1b);
    // ---- FN = LN1 Wf^T + bf ----
    {
        dim3 grid(EMB / BN, SEQ / BM, 1);
        gemm_wmma_bf16<false, true><<<grid, blk, 0, stream>>>(
            LN1b, Wfb, FNf, nullptr, Wf_b,
            SEQ, EMB, EMB, EMB, EMB, EMB,
            0LL, 0LL, 0LL, 0LL, 1.0f);
    }
    // ---- LN2 = layernorm(FN)*g2+b2 + LN1 -> d_out (f32) ----
    layernorm_res_kernel<<<SEQ, blk, 0, stream>>>(FNf, ln2_g, ln2_b, LN1f,
                                                  (float*)d_out, nullptr);
}
